// x_MultiHeadAttention_38689065402383
// MI455X (gfx1250) — matrix operand order repaired and checked by reading, not compiled, not run on hardware
//
#include <hip/hip_runtime.h>
#include <cstdint>

// Problem constants (match reference)
#define B_ 2
#define L_ 2048
#define D_ 1024
#define H_ 16
#define DH_ 64
#define R_ 32
#define BL (B_ * L_)
#define LORA_SCALE 2.0f

typedef __attribute__((ext_vector_type(16))) __bf16 v16bf;
typedef __attribute__((ext_vector_type(8)))  float v8f;

union ABu { v16bf v; uint32_t u[8]; };

__device__ __forceinline__ uint16_t f2bf(float f) {
  uint32_t u = __float_as_uint(f);
  u += 0x7FFFu + ((u >> 16) & 1u);   // round-to-nearest-even
  return (uint16_t)(u >> 16);
}
__device__ __forceinline__ float bf2f(uint16_t h) {
  return __uint_as_float(((uint32_t)h) << 16);
}
__device__ __forceinline__ v8f vzero8() {
  v8f z = {0.f, 0.f, 0.f, 0.f, 0.f, 0.f, 0.f, 0.f};
  return z;
}

// Block-wide sum for blockDim.x == 256 (8 waves of 32).
__device__ __forceinline__ float blockReduceSum(float v, float* red) {
  #pragma unroll
  for (int off = 16; off >= 1; off >>= 1) v += __shfl_xor(v, off);
  int wid = threadIdx.x >> 5, lane = threadIdx.x & 31;
  if (lane == 0) red[wid] = v;
  __syncthreads();
  float r = 0.f;
  #pragma unroll
  for (int i = 0; i < 8; ++i) r += red[i];
  __syncthreads();
  return r;
}

// ---------------------------------------------------------------- cvt f32->bf16
__global__ void cvt_kernel(const float* __restrict__ src, uint16_t* __restrict__ dst, int n) {
  int i = blockIdx.x * blockDim.x + threadIdx.x;
  if (i < n) dst[i] = f2bf(src[i]);
}

// ---------------------------------------------------------------- input LayerNorm
__global__ void ln_kernel(const float* __restrict__ x, const float* __restrict__ w,
                          const float* __restrict__ bia, uint16_t* __restrict__ h) {
  __shared__ float red[8];
  int row = blockIdx.x;
  const float* xr = x + (size_t)row * D_;
  float v[4];
  float s = 0.f;
  #pragma unroll
  for (int i = 0; i < 4; ++i) { v[i] = xr[threadIdx.x + i * 256]; s += v[i]; }
  s = blockReduceSum(s, red);
  float mu = s * (1.f / D_);
  float s2 = 0.f;
  #pragma unroll
  for (int i = 0; i < 4; ++i) { float d = v[i] - mu; s2 += d * d; }
  s2 = blockReduceSum(s2, red);
  float rstd = rsqrtf(s2 * (1.f / D_) + 1e-5f);
  #pragma unroll
  for (int i = 0; i < 4; ++i) {
    int d = threadIdx.x + i * 256;
    h[(size_t)row * D_ + d] = f2bf((v[i] - mu) * rstd * w[d] + bia[d]);
  }
}

// ---------------------------------------------------------------- WMMA NT GEMM
// C[M,N] (f32) = A[M,K] (bf16, K-contig) * B[N,K]^T (bf16, K-contig).
// Block = 8 waves = 128(M) x 64(N) tile; each wave owns a 16x64 strip.
// The 64N x 32K bf16 B panel (4KB) is staged into LDS with
// global_load_async_to_lds_b128 (one 16B chunk per thread), double-buffered
// and tracked with ASYNCcnt, so 8 waves share every B fetch; B operands are
// then 2x ds_load_b128 per tile.
// A-operand layout (16x32 bf16): lane m=l&15; VGPR j holds K pair at
//   2*j + (lane>=16 ? 16 : 0).
// B-operand layout (32x16 bf16): lane n=l&15; VGPR j holds K pair at
//   2*j + (lane>=16 ? 16 : 0).
__global__ void __launch_bounds__(256)
gemm_bf16_nt(const uint16_t* __restrict__ A, const uint16_t* __restrict__ Bm,
             float* __restrict__ C, int M, int N, int K) {
  __shared__ __align__(16) uint16_t Bsh[2][64 * 32];
  const int tid = threadIdx.x;
  const int lane = tid & 31;
  const int hi = lane >> 4;
  const int wlocal = tid >> 5;
  const int ntiles = N >> 6;
  const int bm = blockIdx.x / ntiles;
  const int bn = blockIdx.x % ntiles;
  const int m0 = bm * 128 + wlocal * 16;
  const int n0 = bn * 64;
  const int arow = m0 + (lane & 15);
  if (m0 >= M) return;

  // Async staging assignment: thread copies 16B = panel row tid>>2, chunk tid&3.
  const int srow = tid >> 2;
  const int schunk = (tid & 3) * 8;  // bf16 elements
  const uint16_t* gsrc = Bm + (size_t)(n0 + srow) * K + schunk;
  const uint32_t lds0 = (uint32_t)(uintptr_t)&Bsh[0][srow * 32 + schunk];
  const uint32_t lds1 = (uint32_t)(uintptr_t)&Bsh[1][srow * 32 + schunk];

  v8f acc[4];
  #pragma unroll
  for (int t = 0; t < 4; ++t) acc[t] = vzero8();

  {  // prefetch first panel into buffer 0
    uint64_t ga = (uint64_t)(uintptr_t)gsrc;
    asm volatile("global_load_async_to_lds_b128 %0, %1, off"
                 :: "v"(lds0), "v"(ga) : "memory");
  }

  int buf = 0;
  for (int kb = 0; kb < K; kb += 32) {
    const bool has_next = (kb + 32) < K;
    if (has_next) {
      uint64_t ga = (uint64_t)(uintptr_t)(gsrc + kb + 32);
      uint32_t la = (buf ^ 1) ? lds1 : lds0;
      asm volatile("global_load_async_to_lds_b128 %0, %1, off"
                   :: "v"(la), "v"(ga) : "memory");
      asm volatile("s_wait_asynccnt 0x1" ::: "memory");  // current panel done
    } else {
      asm volatile("s_wait_asynccnt 0x0" ::: "memory");
    }
    __syncthreads();  // panel visible to all 8 waves

    ABu a;
    #pragma unroll
    for (int j = 0; j < 8; ++j) {
      int kk = kb + (j << 1) + (hi << 4);
      a.u[j] = *(const uint32_t*)(A + (size_t)arow * K + kk);
    }
    const uint16_t* bbase = &Bsh[buf][hi * 16];
    #pragma unroll
    for (int t = 0; t < 4; ++t) {
      ABu b;
      const uint32_t* bp = (const uint32_t*)(bbase + (t * 16 + (lane & 15)) * 32);
      #pragma unroll
      for (int j = 0; j < 8; ++j) b.u[j] = bp[j];
      acc[t] = __builtin_amdgcn_wmma_f32_16x16x32_bf16(
          false, a.v, false, b.v, (short)0, acc[t], false, false);
    }
    __syncthreads();  // all reads done before buffer is overwritten
    buf ^= 1;
  }

  // C layout: VGPR r, lane l -> row r + (l>=16)*8, col l&15
  const int r0 = m0 + hi * 8;
  #pragma unroll
  for (int t = 0; t < 4; ++t)
    #pragma unroll
    for (int r = 0; r < 8; ++r)
      C[(size_t)(r0 + r) * N + n0 + t * 16 + (lane & 15)] = acc[t][r];
}

// ---------------------------------------------------------------- LoRA rank-32 down-proj
__global__ void lora_t_kernel(const uint16_t* __restrict__ h,
                              const float* __restrict__ qa, const float* __restrict__ va,
                              float* __restrict__ tq, float* __restrict__ tv) {
  __shared__ float sq[8][32], sv[8][32];
  int row = blockIdx.x;
  int r = threadIdx.x & 31, seg = threadIdx.x >> 5;
  const uint16_t* hr = h + (size_t)row * D_;
  float aq = 0.f, av = 0.f;
  for (int d = seg * 128; d < seg * 128 + 128; ++d) {
    float hv = bf2f(hr[d]);
    aq += hv * qa[r * D_ + d];
    av += hv * va[r * D_ + d];
  }
  sq[seg][r] = aq; sv[seg][r] = av;
  __syncthreads();
  if (threadIdx.x < 32) {
    float a = 0.f, v2 = 0.f;
    #pragma unroll
    for (int s = 0; s < 8; ++s) { a += sq[s][threadIdx.x]; v2 += sv[s][threadIdx.x]; }
    tq[row * R_ + threadIdx.x] = a;
    tv[row * R_ + threadIdx.x] = v2;
  }
}

// ---------------------------------------------------------------- LoRA-add + Q/K LN + RoPE
// Writes Q,K as bf16 [B,H,L,DH]; V as bf16 transposed [B,H,DH,L] (so P*V GEMM is NT).
__global__ void fuse_kernel(const float* __restrict__ qkv,
                            const float* __restrict__ tq, const float* __restrict__ tv,
                            const float* __restrict__ qlb, const float* __restrict__ vlb,
                            const float* __restrict__ qw, const float* __restrict__ kw,
                            uint16_t* __restrict__ Qh, uint16_t* __restrict__ Kh,
                            uint16_t* __restrict__ Vt) {
  __shared__ float qs[D_], ks2[D_], red[8], tqs[R_], tvs[R_];
  int row = blockIdx.x;
  int b = row / L_, l = row % L_;
  if (threadIdx.x < R_) {
    tqs[threadIdx.x] = tq[row * R_ + threadIdx.x];
    tvs[threadIdx.x] = tv[row * R_ + threadIdx.x];
  }
  __syncthreads();
  const float* qr = qkv + (size_t)row * (3 * D_);
  float qreg[4], kreg[4], vv[4];
  float sq_ = 0.f, sk_ = 0.f;
  #pragma unroll
  for (int i = 0; i < 4; ++i) {
    int d = threadIdx.x + i * 256;
    float accq = 0.f, accv = 0.f;
    #pragma unroll
    for (int rr = 0; rr < R_; ++rr) {
      accq += tqs[rr] * qlb[d * R_ + rr];
      accv += tvs[rr] * vlb[d * R_ + rr];
    }
    qreg[i] = qr[d] + LORA_SCALE * accq;
    kreg[i] = qr[D_ + d];
    vv[i]   = qr[2 * D_ + d] + LORA_SCALE * accv;
    qs[d] = qreg[i]; ks2[d] = kreg[i];
    sq_ += qreg[i]; sk_ += kreg[i];
  }
  float muq = blockReduceSum(sq_, red) * (1.f / D_);
  float muk = blockReduceSum(sk_, red) * (1.f / D_);
  float vq = 0.f, vk = 0.f;
  #pragma unroll
  for (int i = 0; i < 4; ++i) {
    float dq = qreg[i] - muq; vq += dq * dq;
    float dk = kreg[i] - muk; vk += dk * dk;
  }
  float rsq2 = rsqrtf(blockReduceSum(vq, red) * (1.f / D_) + 1e-5f);
  float rsk  = rsqrtf(blockReduceSum(vk, red) * (1.f / D_) + 1e-5f);
  #pragma unroll
  for (int i = 0; i < 4; ++i) {
    int d = threadIdx.x + i * 256;
    int hh = d >> 6, dh = d & 63, fi = dh & 31;
    float ang = (float)l * __powf(10000.f, -((float)(2 * fi)) / 64.f);
    float c, s;
    __sincosf(ang, &s, &c);
    float qn = (qreg[i] - muq) * rsq2 * qw[d];
    float kn = (kreg[i] - muk) * rsk * kw[d];
    int pd = (dh < 32) ? d + 32 : d - 32;
    float qp = (qs[pd] - muq) * rsq2 * qw[pd];
    float kp = (ks2[pd] - muk) * rsk * kw[pd];
    float qo = (dh < 32) ? (qn * c - qp * s) : (qn * c + qp * s);
    float ko = (dh < 32) ? (kn * c - kp * s) : (kn * c + kp * s);
    size_t base = ((size_t)(b * H_ + hh) * L_ + l) * DH_ + dh;
    Qh[base] = f2bf(qo);
    Kh[base] = f2bf(ko);
    Vt[((size_t)(b * H_ + hh) * DH_ + dh) * L_ + l] = f2bf(vv[i]);
  }
}

// ---------------------------------------------------------------- flash attention
// One wave per (b, h, 16-query tile). Scores: 2x WMMA (DH=64 = 2 x K32) per
// 16-key tile; online softmax over 32-key steps; P staged bf16 through LDS to
// re-layout C-format -> A-operand; P*V: 4x WMMA per step (N=64).
__global__ void attn_kernel(const uint16_t* __restrict__ Qg,
                            const uint16_t* __restrict__ Kg,
                            const uint16_t* __restrict__ Vg,
                            const int* __restrict__ seq,
                            uint16_t* __restrict__ ctx) {
  __shared__ uint16_t Plds[8][16 * 32];
  const int lane = threadIdx.x & 31;
  const int hi = lane >> 4;
  const int wlocal = threadIdx.x >> 5;
  const int wave = blockIdx.x * (blockDim.x >> 5) + wlocal;
  const int qtiles = L_ / 16;
  const int bh = wave / qtiles;
  if (bh >= B_ * H_) return;
  const int qb = (wave % qtiles) * 16;
  const int b = bh / H_;
  const int hd = bh % H_;
  const uint16_t* Qp = Qg + (size_t)bh * L_ * DH_;
  const uint16_t* Kp = Kg + (size_t)bh * L_ * DH_;
  const uint16_t* Vp = Vg + (size_t)bh * DH_ * L_;
  const int* sq_base = seq + b * L_;

  // Q tile as two A-operands (d 0..31 and 32..63)
  ABu A0, A1;
  {
    const int qrow = qb + (lane & 15);
    #pragma unroll
    for (int j = 0; j < 8; ++j) {
      int kd = (j << 1) + (hi << 4);
      A0.u[j] = *(const uint32_t*)(Qp + (size_t)qrow * DH_ + kd);
      A1.u[j] = *(const uint32_t*)(Qp + (size_t)qrow * DH_ + 32 + kd);
    }
  }
  int sqr[8];
  #pragma unroll
  for (int r = 0; r < 8; ++r) sqr[r] = sq_base[qb + r + hi * 8];

  float mrow[8], lrow[8];
  #pragma unroll
  for (int r = 0; r < 8; ++r) { mrow[r] = -1e30f; lrow[r] = 0.f; }
  v8f O[4];
  #pragma unroll
  for (int t = 0; t < 4; ++t) O[t] = vzero8();

  const float scale = 0.125f;  // 1/sqrt(64)

  for (int kb = 0; kb < L_; kb += 32) {
    v8f S[2];
    #pragma unroll
    for (int t = 0; t < 2; ++t) {
      const int key = kb + t * 16 + (lane & 15);
      ABu B0, B1;
      #pragma unroll
      for (int j = 0; j < 8; ++j) {
        B0.u[j] = *(const uint32_t*)(Kp + (size_t)key * DH_ + 2 * j + (hi << 4));
        B1.u[j] = *(const uint32_t*)(Kp + (size_t)key * DH_ + 32 + 2 * j + (hi << 4));
      }
      S[t] = __builtin_amdgcn_wmma_f32_16x16x32_bf16(
          false, A0.v, false, B0.v, (short)0, vzero8(), false, false);
      S[t] = __builtin_amdgcn_wmma_f32_16x16x32_bf16(
          false, A1.v, false, B1.v, (short)0, S[t], false, false);
    }
    const int ks0 = sq_base[kb + (lane & 15)];
    const int ks1 = sq_base[kb + 16 + (lane & 15)];
    #pragma unroll
    for (int r = 0; r < 8; ++r) {
      bool ok0 = (ks0 == sqr[r]);
      bool ok1 = (ks1 == sqr[r]);
      float s0 = ok0 ? S[0][r] * scale : -1e30f;
      float s1 = ok1 ? S[1][r] * scale : -1e30f;
      float tmax = fmaxf(s0, s1);
      #pragma unroll
      for (int off = 8; off >= 1; off >>= 1) tmax = fmaxf(tmax, __shfl_xor(tmax, off));
      float mnew = fmaxf(mrow[r], tmax);
      float p0 = ok0 ? __expf(s0 - mnew) : 0.f;
      float p1 = ok1 ? __expf(s1 - mnew) : 0.f;
      float rs = p0 + p1;
      #pragma unroll
      for (int off = 8; off >= 1; off >>= 1) rs += __shfl_xor(rs, off);
      float alpha = __expf(mrow[r] - mnew);
      lrow[r] = lrow[r] * alpha + rs;
      mrow[r] = mnew;
      O[0][r] *= alpha; O[1][r] *= alpha; O[2][r] *= alpha; O[3][r] *= alpha;
      Plds[wlocal][(r + hi * 8) * 32 + (lane & 15)] = f2bf(p0);
      Plds[wlocal][(r + hi * 8) * 32 + 16 + (lane & 15)] = f2bf(p1);
    }
    asm volatile("s_wait_dscnt 0x0" ::: "memory");
    ABu Pa;
    {
      const int m = lane & 15;
      #pragma unroll
      for (int j = 0; j < 8; ++j) {
        int kk = (j << 1) + (hi << 4);
        Pa.u[j] = *(const uint32_t*)&Plds[wlocal][m * 32 + kk];
      }
    }
    #pragma unroll
    for (int t = 0; t < 4; ++t) {
      ABu Bv;
      const int dcol = t * 16 + (lane & 15);
      #pragma unroll
      for (int j = 0; j < 8; ++j)
        Bv.u[j] = *(const uint32_t*)(Vp + (size_t)dcol * L_ + kb + 2 * j + (hi << 4));
      O[t] = __builtin_amdgcn_wmma_f32_16x16x32_bf16(
          false, Pa.v, false, Bv.v, (short)0, O[t], false, false);
    }
  }

  #pragma unroll
  for (int r = 0; r < 8; ++r) {
    float inv = (lrow[r] > 0.f) ? 1.f / lrow[r] : 0.f;
    const size_t grow = (size_t)(b * L_ + qb + r + hi * 8);
    #pragma unroll
    for (int t = 0; t < 4; ++t)
      ctx[grow * D_ + hd * DH_ + t * 16 + (lane & 15)] = f2bf(O[t][r] * inv);
  }
}

// ---------------------------------------------------------------- launch
extern "C" void kernel_launch(void* const* d_in, const int* in_sizes, int n_in,
                              void* d_out, int out_size, void* d_ws, size_t ws_size,
                              hipStream_t stream) {
  (void)in_sizes; (void)n_in; (void)out_size; (void)ws_size;
  const float* x     = (const float*)d_in[0];
  const int*   seq   = (const int*)d_in[1];
  const float* ln_w  = (const float*)d_in[2];
  const float* ln_b  = (const float*)d_in[3];
  const float* w_qkv = (const float*)d_in[4];
  const float* q_la  = (const float*)d_in[5];
  const float* q_lb  = (const float*)d_in[6];
  const float* v_la  = (const float*)d_in[7];
  const float* v_lb  = (const float*)d_in[8];
  const float* q_lnw = (const float*)d_in[9];
  const float* k_lnw = (const float*)d_in[10];
  const float* w_out = (const float*)d_in[11];
  float* out = (float*)d_out;

  char* base = (char*)d_ws;
  size_t off = 0;
  auto alloc = [&](size_t bytes) {
    char* p = base + off;
    off = (off + bytes + 255) & ~(size_t)255;
    return p;
  };
  uint16_t* h_bf    = (uint16_t*)alloc((size_t)BL * D_ * 2);
  uint16_t* wqkv_bf = (uint16_t*)alloc((size_t)3 * D_ * D_ * 2);
  uint16_t* wout_bf = (uint16_t*)alloc((size_t)D_ * D_ * 2);
  float*    qkv     = (float*)alloc((size_t)BL * 3 * D_ * 4);
  float*    tq      = (float*)alloc((size_t)BL * R_ * 4);
  float*    tv      = (float*)alloc((size_t)BL * R_ * 4);
  uint16_t* Qh      = (uint16_t*)alloc((size_t)BL * D_ * 2);
  uint16_t* Kh      = (uint16_t*)alloc((size_t)BL * D_ * 2);
  uint16_t* Vt      = (uint16_t*)alloc((size_t)BL * D_ * 2);
  uint16_t* ctx     = (uint16_t*)alloc((size_t)BL * D_ * 2);

  cvt_kernel<<<(3 * D_ * D_ + 255) / 256, 256, 0, stream>>>(w_qkv, wqkv_bf, 3 * D_ * D_);
  cvt_kernel<<<(D_ * D_ + 255) / 256, 256, 0, stream>>>(w_out, wout_bf, D_ * D_);
  ln_kernel<<<BL, 256, 0, stream>>>(x, ln_w, ln_b, h_bf);

  {  // qkv = h @ Wqkv^T : M=4096, N=3072, K=1024 ; block = 128M x 64N
    int blocks = (BL / 128) * ((3 * D_) / 64);
    gemm_bf16_nt<<<blocks, 256, 0, stream>>>(h_bf, wqkv_bf, qkv, BL, 3 * D_, D_);
  }
  lora_t_kernel<<<BL, 256, 0, stream>>>(h_bf, q_la, v_la, tq, tv);
  fuse_kernel<<<BL, 256, 0, stream>>>(qkv, tq, tv, q_lb, v_lb, q_lnw, k_lnw, Qh, Kh, Vt);
  attn_kernel<<<(B_ * H_ * (L_ / 16)) * 32 / 256, 256, 0, stream>>>(Qh, Kh, Vt, seq, ctx);
  {  // out = ctx @ Wout^T : M=4096, N=1024, K=1024
    int blocks = (BL / 128) * (D_ / 64);
    gemm_bf16_nt<<<blocks, 256, 0, stream>>>(ctx, wout_bf, out, BL, D_, D_);
  }
}
